// LSHAttention_64613488001095
// MI455X (gfx1250) — matrix operand order repaired and checked by reading, not compiled, not run on hardware
//
#include <hip/hip_runtime.h>
#include <hip/hip_bf16.h>
#include <math.h>

typedef __attribute__((ext_vector_type(16))) _Float16 v16h;
typedef __attribute__((ext_vector_type(8)))  float    v8f;

// Problem constants (from reference setup_inputs)
constexpr int BATCH = 16;
constexpr int SEQ   = 4096;
constexpr int DIM   = 64;
constexpr int NH    = 8;     // n_hashes
constexpr int NB    = 64;    // n_buckets per hash
constexpr int BS    = 64;    // bucket_size
constexpr int NROT  = 32;    // n_buckets / 2
constexpr int HT    = NH * SEQ;        // 32768 (flattened h*t axis per batch)
constexpr int NCHUNK = NH * NB;        // 512 chunks per batch
constexpr int GB    = NH * NB;         // 512 global bucket ids per batch

// LDS layout for the attention kernel (dynamic shared, base offset 0):
constexpr int LDS_Q     = 0;                       //  8192 B  f16 64x64
constexpr int LDS_K     = 8192;                    // 16384 B  f16 128x64 (normalized)
constexpr int LDS_VT    = 24576;                   // 16384 B  f16 64x128 (V transposed)
constexpr int LDS_DOTS  = 40960;                   // 32768 B  f32 64x128
constexpr int LDS_STGQK = 40960;                   // 32768 B  f32 staging (aliases DOTS)
constexpr int LDS_STGV  = 73728;                   // 32768 B  f32 staging for V
constexpr int LDS_POSQ  = 106496;                  //   256 B
constexpr int LDS_POSK  = 106752;                  //   512 B
constexpr int LDS_TOTAL = 107264;
// probs f16 64x128 aliases [0, 16384) (Q/K dead after phase 1)

// ---------------------------------------------------------------------------
// Kernel 1: LSH hashing. rotated = qk . rotations ; bucket = argmax([r,-r]) + h*NB
// ---------------------------------------------------------------------------
__global__ __launch_bounds__(256) void lsh_hash_kernel(
    const float* __restrict__ qk, const float* __restrict__ rot,
    int* __restrict__ buckets)
{
  __shared__ float rotl[DIM * NROT];   // 8KB: rotations for this block's hash
  const int bid = blockIdx.x;
  const int tb  = bid & 15;            // 16 blocks per (b,h)
  const int h   = (bid >> 4) & 7;
  const int b   = bid >> 7;
  const int tid = threadIdx.x;

  for (int idx = tid; idx < DIM * NROT; idx += 256) {
    int f = idx / NROT, i = idx % NROT;
    rotl[idx] = rot[f * (NH * NROT) + h * NROT + i];
  }
  __syncthreads();

  const int t = tb * 256 + tid;
  const float* qrow = qk + ((size_t)b * SEQ + t) * DIM;
  float q[DIM];
#pragma unroll
  for (int f = 0; f < DIM; ++f) q[f] = qrow[f];

  float r[NROT];
#pragma unroll 4
  for (int i = 0; i < NROT; ++i) {
    float acc = 0.f;
#pragma unroll
    for (int f = 0; f < DIM; ++f) acc += q[f] * rotl[f * NROT + i];
    r[i] = acc;
  }
  float bestv = r[0];
  int besti = 0;
  for (int i = 1; i < NROT; ++i) { if (r[i]  > bestv) { bestv = r[i];  besti = i; } }
  for (int i = 0; i < NROT; ++i) { if (-r[i] > bestv) { bestv = -r[i]; besti = NROT + i; } }

  buckets[(size_t)b * HT + h * SEQ + t] = h * NB + besti;
}

// ---------------------------------------------------------------------------
// Kernel 2a: zero counts
// ---------------------------------------------------------------------------
__global__ void zero_kernel(int* __restrict__ p, int n)
{
  int i = blockIdx.x * blockDim.x + threadIdx.x;
  if (i < n) p[i] = 0;
}

// ---------------------------------------------------------------------------
// Kernel 2b: histogram of global bucket ids per batch
// ---------------------------------------------------------------------------
__global__ __launch_bounds__(256) void hist_kernel(
    const int* __restrict__ buckets, int* __restrict__ counts)
{
  int gid = blockIdx.x * blockDim.x + threadIdx.x;   // BATCH*HT threads
  int b = gid / HT;
  int s = gid % HT;
  int q = buckets[(size_t)b * HT + s];
  atomicAdd(&counts[b * GB + q], 1);
}

// ---------------------------------------------------------------------------
// Kernel 2c: exclusive prefix sum over 512 buckets per batch (tiny)
// ---------------------------------------------------------------------------
__global__ void scan_kernel(const int* __restrict__ counts, int* __restrict__ offs)
{
  int b = blockIdx.x;
  if (threadIdx.x == 0) {
    int acc = 0;
    for (int q = 0; q < GB; ++q) {
      offs[b * GB + q] = acc;
      acc += counts[b * GB + q];
    }
  }
}

// ---------------------------------------------------------------------------
// Kernel 2d: stable placement -> reproduces argsort(bucket*SEQ + t) + inverse
// ---------------------------------------------------------------------------
__global__ __launch_bounds__(256) void place_kernel(
    const int* __restrict__ buckets, const int* __restrict__ offs,
    int* __restrict__ sorted_s, int* __restrict__ inv)
{
  int gid = blockIdx.x * blockDim.x + threadIdx.x;   // BATCH*GB threads
  int b = gid / GB;
  int q = gid % GB;
  int h = q / NB;
  int base = offs[b * GB + q];
  const int* brow = buckets + (size_t)b * HT + h * SEQ;
  int cnt = 0;
  for (int t = 0; t < SEQ; ++t) {
    if (brow[t] == q) {
      int j = base + cnt;
      int s = h * SEQ + t;
      sorted_s[(size_t)b * HT + j] = s;
      inv[(size_t)b * HT + s] = j;
      ++cnt;
    }
  }
}

// ---------------------------------------------------------------------------
// Kernel 3: chunked attention. Async-DMA gather into LDS staging, then
// f16 WMMA (f32 accumulate). One 128-thread (4-wave) workgroup per (chunk,b).
// ---------------------------------------------------------------------------
__global__ __launch_bounds__(128) void lsh_attn_kernel(
    const float* __restrict__ qk, const float* __restrict__ v,
    const int* __restrict__ sorted_s,
    float* __restrict__ so, float* __restrict__ slog)
{
  extern __shared__ char smem[];
  _Float16* Q    = (_Float16*)(smem + LDS_Q);
  _Float16* K    = (_Float16*)(smem + LDS_K);
  _Float16* VT   = (_Float16*)(smem + LDS_VT);     // [d][kvrow] 64x128
  float*    DOTS = (float*)(smem + LDS_DOTS);
  float*    SQK  = (float*)(smem + LDS_STGQK);     // staging, aliases DOTS
  float*    SV   = (float*)(smem + LDS_STGV);
  int*      POSQ = (int*)(smem + LDS_POSQ);
  int*      POSK = (int*)(smem + LDS_POSK);
  _Float16* P    = (_Float16*)(smem + LDS_Q);      // probs, aliases Q+K

  const int chunk = blockIdx.x;
  const int b     = blockIdx.y;
  const int tid   = threadIdx.x;
  const int lane  = tid & 31;
  const int wave  = tid >> 5;

  // ---- async gather: each thread DMAs its kv row (f32) straight into LDS ----
  {
    const int kr = tid;                      // 0..127
    const int srcChunk = (kr < BS) ? chunk : (chunk + NCHUNK - 1) % NCHUNK;
    const int s = sorted_s[(size_t)b * HT + srcChunk * BS + (kr & (BS - 1))];
    const int t = s & (SEQ - 1);
    POSK[kr] = t;
    if (kr < BS) POSQ[kr] = t;
    const float* qrow = qk + ((size_t)b * SEQ + t) * DIM;
    const float* vrow = v  + ((size_t)b * SEQ + t) * DIM;
#pragma unroll
    for (int i = 0; i < 16; ++i) {           // 16 x 16B = one 256B row
      unsigned lq = (unsigned)(LDS_STGQK + kr * 256 + i * 16);
      unsigned lv = (unsigned)(LDS_STGV  + kr * 256 + i * 16);
      const float* gq = qrow + i * 4;
      const float* gv = vrow + i * 4;
      asm volatile("global_load_async_to_lds_b128 %0, %1, off"
                   :: "v"(lq), "v"(gq) : "memory");
      asm volatile("global_load_async_to_lds_b128 %0, %1, off"
                   :: "v"(lv), "v"(gv) : "memory");
    }
  }
  asm volatile("s_wait_asynccnt 0x0" ::: "memory");
  __syncthreads();

  // ---- convert out of staging: Q f16, K f16 (L2-normalized), V^T f16 ----
  {
    const int kr = tid;
    float tmp[DIM];
    float ss = 0.f;
#pragma unroll
    for (int d = 0; d < DIM; ++d) { float x = SQK[kr * DIM + d]; tmp[d] = x; ss += x * x; }
    const float invn = 1.0f / fmaxf(sqrtf(ss), 1e-12f);
#pragma unroll
    for (int d = 0; d < DIM; ++d) {
      K[kr * DIM + d] = (_Float16)(tmp[d] * invn);
      if (kr < BS) Q[kr * DIM + d] = (_Float16)tmp[d];
    }
#pragma unroll
    for (int d = 0; d < DIM; ++d) VT[d * 128 + kr] = (_Float16)SV[kr * DIM + d];
  }
  __syncthreads();

  const int row0 = wave * 16;
  const int l15  = lane & 15;
  const int kbA  = (lane < 16) ? 0 : 16;    // A frag: lanes 0-15 K=0..15
  const int kbB  = (lane < 16) ? 0 : 16;    // B frag: lanes 0-15 K=0..15

  // ---- phase 1: dots = Q . K^T * (1/8) ----
  for (int ct = 0; ct < 8; ++ct) {
    v8f acc = {};
    for (int ks = 0; ks < 2; ++ks) {
      v16h a, bb;
      const int arow = row0 + l15;
      const int bcol = ct * 16 + l15;
      const int ka = ks * 32 + kbA;
      const int kb = ks * 32 + kbB;
#pragma unroll
      for (int e = 0; e < 8; ++e) {
        a[e]     = Q[arow * DIM + ka + e];
        a[e + 8] = Q[arow * DIM + ka + 8 + e];
      }
#pragma unroll
      for (int e = 0; e < 16; ++e)
        bb[e] = K[bcol * DIM + kb + e];
      acc = __builtin_amdgcn_wmma_f32_16x16x32_f16(
          false, a, false, bb, (short)0, acc, false, false);
    }
#pragma unroll
    for (int r = 0; r < 8; ++r) {
      const int rr = row0 + r + ((lane < 16) ? 0 : 8);
      const int cc = ct * 16 + l15;
      DOTS[rr * 128 + cc] = acc[r] * 0.125f;
    }
  }
  __syncthreads();

  // ---- softmax with self-position mask, exact logsumexp (f32) ----
  if (tid < BS) {
    const int i = tid;
    const int pq = POSQ[i];
    float m = -3.0e38f;
    for (int j = 0; j < 128; ++j) {
      float d = DOTS[i * 128 + j];
      if (POSK[j] == pq) d = -50000.0f;
      DOTS[i * 128 + j] = d;
      m = fmaxf(m, d);
    }
    float ssum = 0.f;
    for (int j = 0; j < 128; ++j) ssum += __expf(DOTS[i * 128 + j] - m);
    const float lse = m + logf(ssum);
    slog[(size_t)b * HT + chunk * BS + i] = lse;
    const float invs = 1.0f / ssum;
    for (int j = 0; j < 128; ++j)
      P[i * 128 + j] = (_Float16)(__expf(DOTS[i * 128 + j] - m) * invs);
  }
  __syncthreads();

  // ---- phase 2: bo = probs . V  (B frag now contiguous thanks to V^T) ----
  for (int ct = 0; ct < 4; ++ct) {
    v8f acc = {};
    for (int ks = 0; ks < 4; ++ks) {
      v16h a, bb;
      const int arow = row0 + l15;
      const int bcol = ct * 16 + l15;
      const int ka = ks * 32 + kbA;
      const int kb = ks * 32 + kbB;
#pragma unroll
      for (int e = 0; e < 8; ++e) {
        a[e]     = P[arow * 128 + ka + e];
        a[e + 8] = P[arow * 128 + ka + 8 + e];
      }
#pragma unroll
      for (int e = 0; e < 16; ++e)
        bb[e] = VT[bcol * 128 + kb + e];
      acc = __builtin_amdgcn_wmma_f32_16x16x32_f16(
          false, a, false, bb, (short)0, acc, false, false);
    }
#pragma unroll
    for (int r = 0; r < 8; ++r) {
      const int rr = row0 + r + ((lane < 16) ? 0 : 8);
      const int cc = ct * 16 + l15;
      so[((size_t)b * HT + chunk * BS + rr) * DIM + cc] = acc[r];
    }
  }
}

// ---------------------------------------------------------------------------
// Kernel 4: unsort + combine hash rounds weighted by softmax of logsumexps
// ---------------------------------------------------------------------------
__global__ __launch_bounds__(256) void combine_kernel(
    const float* __restrict__ so, const float* __restrict__ slog,
    const int* __restrict__ inv, float* __restrict__ out)
{
  int gid = blockIdx.x * blockDim.x + threadIdx.x;   // BATCH*SEQ*DIM
  if (gid >= BATCH * SEQ * DIM) return;
  const int d = gid & (DIM - 1);
  const int t = (gid >> 6) & (SEQ - 1);
  const int b = gid >> 18;

  int   j[NH];
  float l[NH];
  float m = -3.0e38f;
#pragma unroll
  for (int h = 0; h < NH; ++h) {
    j[h] = inv[(size_t)b * HT + h * SEQ + t];
    l[h] = slog[(size_t)b * HT + j[h]];
    m = fmaxf(m, l[h]);
  }
  float z = 0.f;
#pragma unroll
  for (int h = 0; h < NH; ++h) z += __expf(l[h] - m);
  const float invz = 1.0f / z;
  float acc = 0.f;
#pragma unroll
  for (int h = 0; h < NH; ++h)
    acc += __expf(l[h] - m) * invz * so[((size_t)b * HT + j[h]) * DIM + d];
  out[gid] = acc;
}

// ---------------------------------------------------------------------------
// Host launcher
// ---------------------------------------------------------------------------
extern "C" void kernel_launch(void* const* d_in, const int* in_sizes, int n_in,
                              void* d_out, int out_size, void* d_ws, size_t ws_size,
                              hipStream_t stream) {
  const float* qk  = (const float*)d_in[0];
  const float* v   = (const float*)d_in[1];
  const float* rot = (const float*)d_in[2];
  float* out = (float*)d_out;

  char* ws = (char*)d_ws;
  size_t off = 0;
  auto carve = [&](size_t bytes) { void* p = ws + off; off += (bytes + 255) & ~(size_t)255; return p; };
  int*   buckets  = (int*)  carve((size_t)BATCH * HT * 4);        // 2 MB
  int*   counts   = (int*)  carve((size_t)BATCH * GB * 4);        // 32 KB
  int*   offs     = (int*)  carve((size_t)BATCH * GB * 4);        // 32 KB
  int*   sorted_s = (int*)  carve((size_t)BATCH * HT * 4);        // 2 MB
  int*   inv      = (int*)  carve((size_t)BATCH * HT * 4);        // 2 MB
  float* slog     = (float*)carve((size_t)BATCH * HT * 4);        // 2 MB
  float* so       = (float*)carve((size_t)BATCH * HT * DIM * 4);  // 128 MB
  (void)ws_size; (void)n_in; (void)in_sizes; (void)out_size;

  // 1) hash
  lsh_hash_kernel<<<BATCH * NH * (SEQ / 256), 256, 0, stream>>>(qk, rot, buckets);

  // 2) stable counting sort by (bucket, position)
  zero_kernel<<<(BATCH * GB + 255) / 256, 256, 0, stream>>>(counts, BATCH * GB);
  hist_kernel<<<(BATCH * HT) / 256, 256, 0, stream>>>(buckets, counts);
  scan_kernel<<<BATCH, 64, 0, stream>>>(counts, offs);
  place_kernel<<<(BATCH * GB) / 256, 256, 0, stream>>>(buckets, offs, sorted_s, inv);

  // 3) chunked WMMA attention with async-to-LDS gather
  lsh_attn_kernel<<<dim3(NCHUNK, BATCH), 128, LDS_TOTAL, stream>>>(qk, v, sorted_s, so, slog);

  // 4) unsort + combine across hash rounds
  combine_kernel<<<(BATCH * SEQ * DIM + 255) / 256, 256, 0, stream>>>(so, slog, inv, out);
}
